// EGNN_41729902247986
// MI455X (gfx1250) — matrix operand order repaired and checked by reading, not compiled, not run on hardware
//
#include <hip/hip_runtime.h>

typedef _Float16 half_t;
typedef __attribute__((ext_vector_type(16))) _Float16 v16h;
typedef __attribute__((ext_vector_type(8)))  _Float16 v8h;
typedef __attribute__((ext_vector_type(8)))  float    v8f;

#define HID 128
#define NLAYERS 4
#define IN_NODE 11
#define NGRAPH 128
#define NPG 64
#define NTOT (NGRAPH * NPG)            // 8192 nodes
#define NEDGE (NTOT * (NPG - 1))       // 516096 edges

#define KT1 9                          // 288 = 9*32 padded K for GEMM1
#define KT2 4                          // 128 = 4*32 for GEMM2
#define NT  8                          // 128 = 8*16 output cols
#define EPB 64                         // edges per block
#define EIN_STRIDE 296                 // halves; 592B row, 16B aligned
#define MID_STRIDE 136                 // halves; 272B row, 16B aligned

__device__ __forceinline__ float silu_f(float x) {
    return x / (1.0f + __expf(-x));
}

// ---------------- weight packing into WMMA B-fragment layout -----------------
// layout: [l][kt][nt][lane(32)][elem(16)]; lane: n=lane&15, khalf=lane>>4;
// elem e -> K = kt*32 + (e<8 ? khalf*8 + e : 16 + khalf*8 + (e-8)) ; N = nt*16 + (lane&15)
__global__ __launch_bounds__(256) void pack_w1_kernel(const float* __restrict__ ew1,
                                                      half_t* __restrict__ w1p) {
    const int idx = blockIdx.x * blockDim.x + threadIdx.x;   // NLAYERS*9*8*32*16
    const int e    = idx & 15;
    const int lane = (idx >> 4) & 31;
    const int nt   = (idx >> 9) & 7;
    const int rem  = idx >> 12;                 // l*9 + kt
    const int kt   = rem % 9;
    const int l    = rem / 9;
    const int k = kt * 32 + (e < 8 ? (lane >> 4) * 8 + e : 16 + (lane >> 4) * 8 + (e - 8));
    const int n = nt * 16 + (lane & 15);
    float v = 0.0f;
    if (k < 261) v = ew1[((size_t)l * 261 + k) * HID + n];
    w1p[idx] = (half_t)v;
}

__global__ __launch_bounds__(256) void pack_w2_kernel(const float* __restrict__ ew2,
                                                      half_t* __restrict__ w2p) {
    const int idx = blockIdx.x * blockDim.x + threadIdx.x;   // NLAYERS*4*8*32*16
    const int e    = idx & 15;
    const int lane = (idx >> 4) & 31;
    const int nt   = (idx >> 9) & 7;
    const int kt   = (idx >> 12) & 3;
    const int l    = idx >> 14;
    const int k = kt * 32 + (e < 8 ? (lane >> 4) * 8 + e : 16 + (lane >> 4) * 8 + (e - 8));
    const int n = nt * 16 + (lane & 15);
    w2p[idx] = (half_t)ew2[((size_t)l * HID + k) * HID + n];
}

// ---------------- embedding: h = h0 @ emb_w + emb_b --------------------------
__global__ __launch_bounds__(256) void embed_kernel(const float* __restrict__ h0,
                                                    const float* __restrict__ emb_w,
                                                    const float* __restrict__ emb_b,
                                                    float* __restrict__ h,
                                                    half_t* __restrict__ hq) {
    const int i = blockIdx.x * blockDim.x + threadIdx.x;     // NTOT*HID
    const int n = i >> 7;
    const int t = i & 127;
    float acc = emb_b[t];
#pragma unroll
    for (int k = 0; k < IN_NODE; ++k)
        acc += h0[n * IN_NODE + k] * emb_w[k * HID + t];
    h[i]  = acc;
    hq[i] = (half_t)acc;
}

// ---------------- radial: ||x_i - x_j||^2 ------------------------------------
__global__ __launch_bounds__(256) void radial_kernel(const float* __restrict__ x,
                                                     const int* __restrict__ rows,
                                                     const int* __restrict__ cols,
                                                     float* __restrict__ radial) {
    const int e = blockIdx.x * blockDim.x + threadIdx.x;
    const int r = rows[e], c = cols[e];
    const float dx = x[r * 3 + 0] - x[c * 3 + 0];
    const float dy = x[r * 3 + 1] - x[c * 3 + 1];
    const float dz = x[r * 3 + 2] - x[c * 3 + 2];
    radial[e] = dx * dx + dy * dy + dz * dz;
}

__global__ __launch_bounds__(256) void zero_kernel(float* __restrict__ p) {
    p[blockIdx.x * blockDim.x + threadIdx.x] = 0.0f;
}

// ---------------- fused edge MLP + scatter-sum (WMMA) ------------------------
__global__ __launch_bounds__(256) void egcl_edge_kernel(
    const half_t* __restrict__ hq,       // [NTOT][HID] f16
    const float*  __restrict__ radial,   // [NEDGE]
    const float*  __restrict__ eattr,    // [NEDGE][4]
    const int*    __restrict__ rows,
    const int*    __restrict__ cols,
    const float*  __restrict__ emask,    // [NEDGE]
    const half_t* __restrict__ w1p,      // packed [9][8][32][16]
    const float*  __restrict__ b1,       // [HID]
    const half_t* __restrict__ w2p,      // packed [4][8][32][16]
    const float*  __restrict__ b2,       // [HID]
    float* __restrict__ agg)             // [NTOT][HID]
{
    __shared__ half_t ein[EPB * EIN_STRIDE];
    __shared__ half_t mid[EPB * MID_STRIDE];
    __shared__ float  laggr[2 * HID];
    __shared__ int    srow[EPB];
    __shared__ float  smask[EPB];

    const int tid = threadIdx.x;
    const long eBase = (long)blockIdx.x * EPB;

    laggr[tid] = 0.0f;

    // ---- stage concat[h_row, h_col, radial, edge_attr, pad] as f16 in LDS ----
    {
        const int le   = tid >> 2;      // edge in block
        const int part = tid & 3;
        const long e = eBase + le;
        if (part == 0) { srow[le] = rows[e]; smask[le] = emask[e]; }
        const int node = (part < 2) ? rows[e] : cols[e];
        const uint4* s4 = (const uint4*)(hq + (long)node * HID + (part & 1) * 64);
        uint4* d4 = (uint4*)(&ein[le * EIN_STRIDE + (part & 1) * 64 + ((part < 2) ? 0 : HID)]);
#pragma unroll
        for (int i = 0; i < 8; ++i) d4[i] = s4[i];
        if (part == 3) {
            half_t* t = &ein[le * EIN_STRIDE + 2 * HID];
            t[0] = (half_t)radial[e];
            t[1] = (half_t)eattr[e * 4 + 0];
            t[2] = (half_t)eattr[e * 4 + 1];
            t[3] = (half_t)eattr[e * 4 + 2];
            t[4] = (half_t)eattr[e * 4 + 3];
#pragma unroll
            for (int i = 5; i < 32; ++i) t[i] = (half_t)0.0f;   // K pad 261..287
        }
    }
    __syncthreads();

    const int wave   = tid >> 5;        // 8 waves
    const int lane   = tid & 31;
    const int l15    = lane & 15;       // A: M row ; C: N col
    const int khalf  = lane >> 4;
    const int edge0  = (wave >> 1) * 16;        // 4 M-tiles of 16 edges
    const int ntBase = (wave & 1) * 4;          // each wave: 4 N-tiles (64 cols)

    // ------------------------- GEMM1: [16,288] x [288,128] --------------------
    v8f acc[4] = {};
#pragma unroll
    for (int kt = 0; kt < KT1; ++kt) {
        const half_t* ap = &ein[(edge0 + l15) * EIN_STRIDE + kt * 32 + khalf * 8];
        v8h alo = *(const v8h*)ap;          // K = kt*32 + khalf*8 + 0..7
        v8h ahi = *(const v8h*)(ap + 16);   // K = kt*32 + 16 + khalf*8 + 0..7
        v16h a = __builtin_shufflevector(alo, ahi, 0,1,2,3,4,5,6,7,8,9,10,11,12,13,14,15);
#pragma unroll
        for (int j = 0; j < 4; ++j) {
            const int nt = ntBase + j;
            v16h b = *(const v16h*)(w1p + ((size_t)(kt * NT + nt) * 32 + lane) * 16);
            acc[j] = __builtin_amdgcn_wmma_f32_16x16x32_f16(
                false, a, false, b, (short)0, acc[j], false, false);
        }
    }

    // bias + SiLU, lane-transpose through LDS into A layout for GEMM2
#pragma unroll
    for (int j = 0; j < 4; ++j) {
        const int colIdx = (ntBase + j) * 16 + l15;
        const float bias = b1[colIdx];
#pragma unroll
        for (int v = 0; v < 8; ++v) {
            const float xv = silu_f(acc[j][v] + bias);
            mid[(edge0 + v + 8 * khalf) * MID_STRIDE + colIdx] = (half_t)xv;
        }
    }
    __syncthreads();

    // ------------------------- GEMM2: [16,128] x [128,128] --------------------
    v8f acc2[4] = {};
#pragma unroll
    for (int kt = 0; kt < KT2; ++kt) {
        const half_t* ap = &mid[(edge0 + l15) * MID_STRIDE + kt * 32 + khalf * 8];
        v8h alo = *(const v8h*)ap;
        v8h ahi = *(const v8h*)(ap + 16);
        v16h a = __builtin_shufflevector(alo, ahi, 0,1,2,3,4,5,6,7,8,9,10,11,12,13,14,15);
#pragma unroll
        for (int j = 0; j < 4; ++j) {
            const int nt = ntBase + j;
            v16h b = *(const v16h*)(w2p + ((size_t)(kt * NT + nt) * 32 + lane) * 16);
            acc2[j] = __builtin_amdgcn_wmma_f32_16x16x32_f16(
                false, a, false, b, (short)0, acc2[j], false, false);
        }
    }

    // ---- bias + SiLU + edge_mask, reduce per-row (<=2 rows per block) --------
    const int rowFirst = srow[0];
#pragma unroll
    for (int j = 0; j < 4; ++j) {
        const int colIdx = (ntBase + j) * 16 + l15;
        const float bias = b2[colIdx];
#pragma unroll
        for (int v = 0; v < 8; ++v) {
            const int mle = edge0 + v + 8 * khalf;
            const float xv = silu_f(acc2[j][v] + bias) * smask[mle];
            const int slot = (srow[mle] != rowFirst) ? 1 : 0;
            atomicAdd(&laggr[slot * HID + colIdx], xv);
        }
    }
    __syncthreads();

    if (tid < HID) {
        atomicAdd(&agg[(long)rowFirst * HID + tid], laggr[tid]);
    } else {
        const int rowLast = srow[EPB - 1];
        if (rowLast != rowFirst)
            atomicAdd(&agg[(long)rowLast * HID + (tid - HID)], laggr[HID + (tid - HID)]);
    }
}

// ---------------- node MLP (fp32, residual) ----------------------------------
__global__ __launch_bounds__(128) void node_kernel(
    float* __restrict__ h, const float* __restrict__ agg, const float* __restrict__ h0,
    const float* __restrict__ w1, const float* __restrict__ b1,
    const float* __restrict__ w2, const float* __restrict__ b2,
    half_t* __restrict__ hq)
{
    __shared__ float nin[2 * HID + IN_NODE];
    __shared__ float hidb[HID];
    const int n = blockIdx.x;
    const int t = threadIdx.x;
    nin[t]       = h[(size_t)n * HID + t];
    nin[HID + t] = agg[(size_t)n * HID + t];
    if (t < IN_NODE) nin[2 * HID + t] = h0[(size_t)n * IN_NODE + t];
    __syncthreads();
    float acc = b1[t];
    for (int k = 0; k < 2 * HID + IN_NODE; ++k) acc += nin[k] * w1[(size_t)k * HID + t];
    hidb[t] = silu_f(acc);
    __syncthreads();
    float acc2 = b2[t];
    for (int k = 0; k < HID; ++k) acc2 += hidb[k] * w2[(size_t)k * HID + t];
    const float hv = nin[t] + acc2;
    h[(size_t)n * HID + t]  = hv;
    hq[(size_t)n * HID + t] = (half_t)hv;
}

// ---------------- node decoder + masked pool + graph decoder -----------------
__global__ __launch_bounds__(128) void decode_kernel(
    const float* __restrict__ h, const float* __restrict__ node_mask,
    const float* __restrict__ dw1, const float* __restrict__ db1,
    const float* __restrict__ dw2, const float* __restrict__ db2,
    const float* __restrict__ gw1, const float* __restrict__ gb1,
    const float* __restrict__ gw2, const float* __restrict__ gb2,
    float* __restrict__ out)
{
    __shared__ float buf[HID];
    __shared__ float hid[HID];
    const int g = blockIdx.x;
    const int t = threadIdx.x;
    float gacc = 0.0f;
    for (int n = 0; n < NPG; ++n) {
        const size_t node = (size_t)g * NPG + n;
        buf[t] = h[node * HID + t];
        __syncthreads();
        float a = db1[t];
        for (int k = 0; k < HID; ++k) a += buf[k] * dw1[(size_t)k * HID + t];
        hid[t] = silu_f(a);
        __syncthreads();
        float a2 = db2[t];
        for (int k = 0; k < HID; ++k) a2 += hid[k] * dw2[(size_t)k * HID + t];
        gacc += a2 * node_mask[node];
        __syncthreads();
    }
    buf[t] = gacc;
    __syncthreads();
    float a = gb1[t];
    for (int k = 0; k < HID; ++k) a += buf[k] * gw1[(size_t)k * HID + t];
    hid[t] = silu_f(a) * gw2[t];
    __syncthreads();
    for (int s = 64; s > 0; s >>= 1) {
        if (t < s) hid[t] += hid[t + s];
        __syncthreads();
    }
    if (t == 0) out[g] = hid[0] + gb2[0];
}

// ============================== launcher =====================================
extern "C" void kernel_launch(void* const* d_in, const int* in_sizes, int n_in,
                              void* d_out, int out_size, void* d_ws, size_t ws_size,
                              hipStream_t stream) {
    (void)in_sizes; (void)n_in; (void)out_size; (void)ws_size;
    const float* h0        = (const float*)d_in[0];
    const float* x         = (const float*)d_in[1];
    const int*   edges     = (const int*)d_in[2];
    const float* edge_attr = (const float*)d_in[3];
    const float* node_mask = (const float*)d_in[4];
    const float* edge_mask = (const float*)d_in[5];
    // d_in[6] = n_nodes (compile-time constant here)
    const float* emb_w  = (const float*)d_in[7];
    const float* emb_b  = (const float*)d_in[8];
    const float* ew1    = (const float*)d_in[9];
    const float* eb1    = (const float*)d_in[10];
    const float* ew2    = (const float*)d_in[11];
    const float* eb2    = (const float*)d_in[12];
    const float* nw1    = (const float*)d_in[13];
    const float* nb1    = (const float*)d_in[14];
    const float* nw2    = (const float*)d_in[15];
    const float* nb2    = (const float*)d_in[16];
    const float* dw1    = (const float*)d_in[17];
    const float* db1    = (const float*)d_in[18];
    const float* dw2    = (const float*)d_in[19];
    const float* db2    = (const float*)d_in[20];
    const float* gw1    = (const float*)d_in[21];
    const float* gb1    = (const float*)d_in[22];
    const float* gw2    = (const float*)d_in[23];
    const float* gb2    = (const float*)d_in[24];
    float* out = (float*)d_out;

    const int* rows = edges;
    const int* cols = edges + NEDGE;

    // workspace carve-up (all 256B-aligned by construction)
    char* ws = (char*)d_ws;
    float*  h      = (float*)ws;   ws += (size_t)NTOT * HID * 4;     // 4 MB
    half_t* hq     = (half_t*)ws;  ws += (size_t)NTOT * HID * 2;     // 2 MB
    float*  agg    = (float*)ws;   ws += (size_t)NTOT * HID * 4;     // 4 MB
    float*  radial = (float*)ws;   ws += (size_t)NEDGE * 4;          // ~2 MB
    half_t* w1p    = (half_t*)ws;  ws += (size_t)NLAYERS * KT1 * NT * 32 * 16 * 2;
    half_t* w2p    = (half_t*)ws;  ws += (size_t)NLAYERS * KT2 * NT * 32 * 16 * 2;

    pack_w1_kernel<<<(NLAYERS * KT1 * NT * 32 * 16) / 256, 256, 0, stream>>>(ew1, w1p);
    pack_w2_kernel<<<(NLAYERS * KT2 * NT * 32 * 16) / 256, 256, 0, stream>>>(ew2, w2p);
    embed_kernel<<<(NTOT * HID) / 256, 256, 0, stream>>>(h0, emb_w, emb_b, h, hq);
    radial_kernel<<<NEDGE / 256, 256, 0, stream>>>(x, rows, cols, radial);

    for (int l = 0; l < NLAYERS; ++l) {
        zero_kernel<<<(NTOT * HID) / 256, 256, 0, stream>>>(agg);
        egcl_edge_kernel<<<NEDGE / EPB, 256, 0, stream>>>(
            hq, radial, edge_attr, rows, cols, edge_mask,
            w1p + (size_t)l * KT1 * NT * 32 * 16, eb1 + l * HID,
            w2p + (size_t)l * KT2 * NT * 32 * 16, eb2 + l * HID, agg);
        node_kernel<<<NTOT, 128, 0, stream>>>(
            h, agg, h0,
            nw1 + (size_t)l * (2 * HID + IN_NODE) * HID, nb1 + l * HID,
            nw2 + (size_t)l * HID * HID, nb2 + l * HID, hq);
    }

    decode_kernel<<<NGRAPH, 128, 0, stream>>>(h, node_mask, dw1, db1, dw2, db2,
                                              gw1, gb1, gw2, gb2, out);
}
